// SphericalConv_16509854286074
// MI455X (gfx1250) — compile-verified, matrix operand order checked by reading, not run on hardware
//
#include <hip/hip_runtime.h>

typedef __attribute__((ext_vector_type(16))) _Float16 v16h;
typedef __attribute__((ext_vector_type(8)))  _Float16 v8h;
typedef __attribute__((ext_vector_type(8)))  float    v8f;

#define Bn    2
#define Nn    16384
#define Cn    64
#define Kn    17
#define Jn    16
#define COn   128
#define MPTS  32   // points per block (two 16-row M-tiles per wave)

// ---------------------------------------------------------------------------
// Kernel 1: repack weight (COUT, C, K-1) fp32  ->  Wh[j][o][c] f16
// Layout chosen so a WMMA B-fragment (K x N = 32 x 16, f16) is, per lane,
// two contiguous 8-halfword (16B) loads: lane = column o, K-values along c.
// ---------------------------------------------------------------------------
__global__ void __launch_bounds__(256)
prep_weight(const float* __restrict__ w, _Float16* __restrict__ wh) {
    int idx = blockIdx.x * blockDim.x + threadIdx.x;   // j*8192 + o*64 + c
    if (idx >= Jn * COn * Cn) return;
    int c = idx & 63;
    int o = (idx >> 6) & 127;
    int j = idx >> 13;
    float v = w[(o * Cn + c) * (Kn - 1) + j];
    wh[idx] = (_Float16)v;
}

// ---------------------------------------------------------------------------
// Kernel 2: main spherical conv. One block = 32 points, 8 waves.
// Each wave owns one 16-wide output coltile and two 16-row point tiles, so
// every B (weight) fragment load feeds two WMMAs.
// ---------------------------------------------------------------------------
__global__ void __launch_bounds__(256)
sph_main(const float* __restrict__ x, const int* __restrict__ adj,
         const _Float16* __restrict__ wh, float* __restrict__ out) {
    // proj staged in LDS as f16: [j][pt][c], pitch 72 halfwords (144B) so the
    // 16 lanes of an A-fragment LDS read land on disjoint bank groups.
    __shared__ _Float16 sproj[Jn][MPTS][72];

    const int tile = blockIdx.x;                 // 0 .. B*N/MPTS - 1
    const int b    = tile / (Nn / MPTS);
    const int nb   = (tile % (Nn / MPTS)) * MPTS; // first point of this tile
    const int t    = threadIdx.x;

    // -------- Phase 1: thread t -> (pt, j) x2; gather + normalize diff -----
    {
        const int j = t & 15;
#pragma unroll
        for (int rep = 0; rep < 2; ++rep) {
            const int pt  = (t >> 4) + rep * 16;
            const int n   = nb + pt;
            const int nbr = adj[((size_t)b * Nn + n) * Kn + 1 + j];
            const float4* xc = (const float4*)(x + ((size_t)b * Nn + n)   * Cn);
            const float4* xn = (const float4*)(x + ((size_t)b * Nn + nbr) * Cn);
            float4 d[16];
            float  ss = 0.0f;
#pragma unroll
            for (int i = 0; i < 16; ++i) {
                float4 a = xn[i], c0 = xc[i], dd;
                dd.x = a.x - c0.x; dd.y = a.y - c0.y;
                dd.z = a.z - c0.z; dd.w = a.w - c0.w;
                ss += dd.x * dd.x + dd.y * dd.y + dd.z * dd.z + dd.w * dd.w;
                d[i] = dd;
            }
            // concat([diff, ||diff||]) renormalized => diff * rsqrt(2*ss)
            const float s = __frsqrt_rn(2.0f * ss);
            _Float16* row = &sproj[j][pt][0];
#pragma unroll
            for (int i = 0; i < 16; ++i) {
                row[4 * i + 0] = (_Float16)(d[i].x * s);
                row[4 * i + 1] = (_Float16)(d[i].y * s);
                row[4 * i + 2] = (_Float16)(d[i].z * s);
                row[4 * i + 3] = (_Float16)(d[i].w * s);
            }
        }
    }
    __syncthreads();

    // -------- Phase 2: wave w owns output coltile o = [16w, 16w+16) --------
    const int wv  = t >> 5;                 // wave id = coltile
    const int L   = t & 31;                 // lane id
    const int hi  = (L < 16) ? 0 : 1;       // lane half (K-split select)
    const int lm  = L & 15;                 // A row (pt) / B column (o) index
    const int o   = wv * 16 + lm;
    const int co  = hi * 8;                 // f16 K-offset for this lane half

    v8f mx0 = {};                           // running max; 0-init folds relu
    v8f mx1 = {};
#pragma unroll
    for (int j = 0; j < Jn; ++j) {
        union { v16h v; v8h h[2]; } A0a, A1a, A0b, A1b, B0, B1;
        // A fragments: point tile 0 (pts 0..15) and tile 1 (pts 16..31)
        const _Float16* ar0 = &sproj[j][lm][0];
        const _Float16* ar1 = &sproj[j][lm + 16][0];
        // k-chunk 0: c in [0,32); lane<16 gets {0..7,16..23}, lane>=16 +8
        A0a.h[0] = *(const v8h*)(ar0 + 0  + co);
        A0a.h[1] = *(const v8h*)(ar0 + 16 + co);
        A1a.h[0] = *(const v8h*)(ar0 + 32 + co);
        A1a.h[1] = *(const v8h*)(ar0 + 48 + co);
        A0b.h[0] = *(const v8h*)(ar1 + 0  + co);
        A0b.h[1] = *(const v8h*)(ar1 + 16 + co);
        A1b.h[0] = *(const v8h*)(ar1 + 32 + co);
        A1b.h[1] = *(const v8h*)(ar1 + 48 + co);

        // B fragments: shared by both point tiles (2x reuse per load)
        const _Float16* brow = wh + ((size_t)j * COn + o) * Cn;
        B0.h[0] = *(const v8h*)(brow + 0  + co);
        B0.h[1] = *(const v8h*)(brow + 16 + co);
        B1.h[0] = *(const v8h*)(brow + 32 + co);
        B1.h[1] = *(const v8h*)(brow + 48 + co);

        v8f acc0 = {};
        acc0 = __builtin_amdgcn_wmma_f32_16x16x32_f16(
                   false, A0a.v, false, B0.v, (short)0, acc0, false, false);
        acc0 = __builtin_amdgcn_wmma_f32_16x16x32_f16(
                   false, A1a.v, false, B1.v, (short)0, acc0, false, false);
        v8f acc1 = {};
        acc1 = __builtin_amdgcn_wmma_f32_16x16x32_f16(
                   false, A0b.v, false, B0.v, (short)0, acc1, false, false);
        acc1 = __builtin_amdgcn_wmma_f32_16x16x32_f16(
                   false, A1b.v, false, B1.v, (short)0, acc1, false, false);
#pragma unroll
        for (int i = 0; i < 8; ++i) {
            mx0[i] = fmaxf(mx0[i], acc0[i]);
            mx1[i] = fmaxf(mx1[i], acc1[i]);
        }
    }

    // -------- Store: lane holds out[b][o][n0 + 0..7] (contiguous) ----------
    union { v8f v; float4 f[2]; } R0, R1;
    R0.v = mx0;
    R1.v = mx1;
    const int n0 = nb + hi * 8;              // point tile 0
    float* op0 = out + ((size_t)b * COn + o) * Nn + n0;
    *(float4*)(op0)     = R0.f[0];
    *(float4*)(op0 + 4) = R0.f[1];
    const int n1 = nb + 16 + hi * 8;         // point tile 1
    float* op1 = out + ((size_t)b * COn + o) * Nn + n1;
    *(float4*)(op1)     = R1.f[0];
    *(float4*)(op1 + 4) = R1.f[1];
}

// ---------------------------------------------------------------------------
extern "C" void kernel_launch(void* const* d_in, const int* in_sizes, int n_in,
                              void* d_out, int out_size, void* d_ws, size_t ws_size,
                              hipStream_t stream) {
    const float* x   = (const float*)d_in[0];
    const int*   adj = (const int*)d_in[1];
    const float* w   = (const float*)d_in[2];
    _Float16*    wh  = (_Float16*)d_ws;          // 16*128*64 f16 = 256 KB
    float*       out = (float*)d_out;

    prep_weight<<<(Jn * COn * Cn + 255) / 256, 256, 0, stream>>>(w, wh);
    sph_main<<<(Bn * Nn) / MPTS, 256, 0, stream>>>(x, adj, wh, out);
}
